// AgendaCopyGenerator_15650860827128
// MI455X (gfx1250) — compile-verified, matrix operand order checked by reading, not run on hardware
//
#include <hip/hip_runtime.h>
#include <hip/hip_bf16.h>
#include <cmath>

// ---------------- CDNA5 vector types ----------------
typedef __attribute__((ext_vector_type(16))) _Float16 v16h;
typedef __attribute__((ext_vector_type(8)))  float    v8f;
typedef __attribute__((ext_vector_type(4)))  float    f32x4;
typedef __attribute__((ext_vector_type(2)))  unsigned int u32x2;

// ---------------- problem constants ----------------
constexpr int N_ROWS  = 2048;            // BATCH * TLEN
constexpr int D       = 1024;            // INPUT_SIZE (K)
constexpr int VOCAB   = 32000;
constexpr int CVOCAB  = 120;
constexpr int SLEN    = 400;
constexpr int AGENDA  = 100;
constexpr int CTX     = SLEN - AGENDA;   // 300
constexpr int BATCH_  = 16;
constexpr int OUTW    = VOCAB + CVOCAB;  // 32120
constexpr int PAD_IDX = 1;

// GEMM tiling
constexpr int BM = 128, BN = 128, BK = 32;
constexpr int LDS_ROW = 40;              // halves per LDS row: 32 data + 8 pad = 80B (16B aligned, bank-friendly)

// ---------------- kernel 1: p_copy = sigmoid(hidden @ W_copy.T + b_copy) ----------------
__global__ __launch_bounds__(256)
void pcopy_kernel(const float* __restrict__ hidden,
                  const float* __restrict__ Wc,
                  const float* __restrict__ bc,
                  float* __restrict__ pcopy_out) {
  const int row = blockIdx.x;
  const int tid = threadIdx.x;
  float s = 0.f;
  for (int k = tid; k < D; k += 256)
    s += hidden[(size_t)row * D + k] * Wc[k];
  __shared__ float red[256];
  red[tid] = s;
  __syncthreads();
  for (int off = 128; off > 0; off >>= 1) {
    if (tid < off) red[tid] += red[tid + off];
    __syncthreads();
  }
  if (tid == 0) {
    float z = red[0] + bc[0];
    pcopy_out[row] = 1.f / (1.f + __expf(-z));
  }
}

// ---------------- kernel 2: logits = hidden @ W.T  (raw, bias added later) ----------------
// A = hidden (M x K) row-major, Bw = W (Nv x K) row-major -> C = A * Bw^T.
// f32 tiles are converted to f16 in-register and staged to LDS; compute uses
// v_wmma_f32_16x16x32_f16 with f32 accumulation. W (131MB) fits in the 192MB L2,
// so re-reads across the 16 M-tiles stay on-chip.
__global__ __launch_bounds__(256)
void gemm_logits_kernel(const float* __restrict__ A,
                        const float* __restrict__ Bw,
                        float* __restrict__ out) {
  __shared__ _Float16 Ash[BM * LDS_ROW];
  __shared__ _Float16 Bsh[BN * LDS_ROW];

  const int tid = threadIdx.x;
  const int m0  = blockIdx.y * BM;
  const int n0  = blockIdx.x * BN;

  const int wave = tid >> 5;       // 8 waves
  const int lane = tid & 31;
  const int wm   = wave & 3;       // 4 waves along M (32 rows each)
  const int wn   = wave >> 2;      // 2 waves along N (64 cols each)
  const int lr   = lane & 15;
  const int hi   = lane >> 4;

  v8f acc[2][4];
#pragma unroll
  for (int i = 0; i < 2; i++)
#pragma unroll
    for (int j = 0; j < 4; j++) acc[i][j] = (v8f)0.f;

  union Frag { v16h h; f32x4 f[2]; };

  for (int k0 = 0; k0 < D; k0 += BK) {
    __syncthreads();
    // stage 128x32 f32 -> f16 LDS tiles (1024 float4 per matrix, 4 per thread)
#pragma unroll
    for (int i = 0; i < 4; i++) {
      const int fid = tid + 256 * i;
      const int row = fid >> 3;
      const int kk  = (fid & 7) * 4;
      f32x4 a = *reinterpret_cast<const f32x4*>(A  + (size_t)(m0 + row) * D + k0 + kk);
      f32x4 b = *reinterpret_cast<const f32x4*>(Bw + (size_t)(n0 + row) * D + k0 + kk);
      union { _Float16 h[4]; u32x2 u; } pa, pb;
      pa.h[0] = (_Float16)a.x; pa.h[1] = (_Float16)a.y;
      pa.h[2] = (_Float16)a.z; pa.h[3] = (_Float16)a.w;
      pb.h[0] = (_Float16)b.x; pb.h[1] = (_Float16)b.y;
      pb.h[2] = (_Float16)b.z; pb.h[3] = (_Float16)b.w;
      *reinterpret_cast<u32x2*>(&Ash[row * LDS_ROW + kk]) = pa.u;
      *reinterpret_cast<u32x2*>(&Bsh[row * LDS_ROW + kk]) = pb.u;
      if (k0 + BK < D) {                        // hint next K-tile into cache
        __builtin_prefetch(A  + (size_t)(m0 + row) * D + k0 + BK + kk, 0, 1);
        __builtin_prefetch(Bw + (size_t)(n0 + row) * D + k0 + BK + kk, 0, 1);
      }
    }
    __syncthreads();

    // fragment loads per ISA 16-bit A layout:
    // lanes 0-15 (hi=0): K 0..7 and 16..23 ; lanes 16-31 (hi=1): K 8..15 and 24..31
    Frag afrag[2], bfrag[4];
#pragma unroll
    for (int im = 0; im < 2; im++) {
      const int r = wm * 32 + im * 16 + lr;
      const _Float16* p = &Ash[r * LDS_ROW + hi * 8];
      afrag[im].f[0] = *reinterpret_cast<const f32x4*>(p);
      afrag[im].f[1] = *reinterpret_cast<const f32x4*>(p + 16);
    }
#pragma unroll
    for (int in = 0; in < 4; in++) {
      const int r = wn * 64 + in * 16 + lr;
      const _Float16* p = &Bsh[r * LDS_ROW + hi * 8];
      bfrag[in].f[0] = *reinterpret_cast<const f32x4*>(p);
      bfrag[in].f[1] = *reinterpret_cast<const f32x4*>(p + 16);
    }

#pragma unroll
    for (int im = 0; im < 2; im++)
#pragma unroll
      for (int in = 0; in < 4; in++)
        acc[im][in] = __builtin_amdgcn_wmma_f32_16x16x32_f16(
            false, afrag[im].h, false, bfrag[in].h,
            (short)0, acc[im][in], false, false);
  }

  // C layout: VGPR r -> M = r + 8*hi, N = lane&15
#pragma unroll
  for (int im = 0; im < 2; im++) {
#pragma unroll
    for (int in = 0; in < 4; in++) {
      const int mbase = m0 + wm * 32 + im * 16 + hi * 8;
      const int n     = n0 + wn * 64 + in * 16 + lr;
#pragma unroll
      for (int r = 0; r < 8; r++)
        out[(size_t)(mbase + r) * OUTW + n] = acc[im][in][r];
    }
  }
}

// ---------------- kernel 3: in-place softmax + PAD mask + (1 - p_copy) scale ----------------
__global__ __launch_bounds__(256)
void softmax_scale_kernel(const float* __restrict__ bias,
                          const float* __restrict__ pcopy,
                          float* __restrict__ out) {
  const int row = blockIdx.x;
  const int tid = threadIdx.x;
  float* lrow = out + (size_t)row * OUTW;

  float mx = -INFINITY, sum = 0.f;
  for (int v = tid; v < VOCAB; v += 256) {
    if (v == PAD_IDX) continue;
    float l = lrow[v] + bias[v];
    if (l > mx) { sum = sum * __expf(mx - l) + 1.f; mx = l; }
    else        { sum += __expf(l - mx); }
  }

  __shared__ float smax[256], ssum[256];
  smax[tid] = mx; ssum[tid] = sum;
  __syncthreads();
  for (int off = 128; off > 0; off >>= 1) {
    if (tid < off) {
      float m1 = smax[tid], s1 = ssum[tid];
      float m2 = smax[tid + off], s2 = ssum[tid + off];
      float M = fmaxf(m1, m2);
      float s = 0.f;
      if (m1 > -INFINITY) s += s1 * __expf(m1 - M);
      if (m2 > -INFINITY) s += s2 * __expf(m2 - M);
      smax[tid] = M; ssum[tid] = s;
    }
    __syncthreads();
  }
  const float M     = smax[0];
  const float scale = (1.f - pcopy[row]) / ssum[0];

  for (int v = tid; v < VOCAB; v += 256) {
    float o = (v == PAD_IDX) ? 0.f : __expf(lrow[v] + bias[v] - M) * scale;
    lrow[v] = o;
  }
}

// ---------------- kernel 4: copy_prob[t,b,:] = p_copy * (attn_agenda row @ src_map[:,b,:]) ----------------
__global__ __launch_bounds__(128)
void copyprob_kernel(const float* __restrict__ attn,
                     const float* __restrict__ srcmap,
                     const float* __restrict__ pcopy,
                     float* __restrict__ out) {
  const int n   = blockIdx.x;          // rows ordered (tlen, batch)
  const int b   = n & (BATCH_ - 1);
  const int tid = threadIdx.x;

  __shared__ float arow[AGENDA];
  for (int a = tid; a < AGENDA; a += blockDim.x)
    arow[a] = attn[(size_t)n * SLEN + CTX + a];
  __syncthreads();

  const float p = pcopy[n];
  for (int v = tid; v < CVOCAB; v += blockDim.x) {
    float s = 0.f;
#pragma unroll 4
    for (int a = 0; a < AGENDA; a++)
      s += arow[a] * srcmap[(size_t)(a * BATCH_ + b) * CVOCAB + v];
    out[(size_t)n * OUTW + VOCAB + v] = s * p;
  }
}

// ---------------- launch ----------------
extern "C" void kernel_launch(void* const* d_in, const int* in_sizes, int n_in,
                              void* d_out, int out_size, void* d_ws, size_t ws_size,
                              hipStream_t stream) {
  (void)in_sizes; (void)n_in; (void)out_size; (void)d_ws; (void)ws_size;

  const float* hidden = (const float*)d_in[0];
  const float* attn   = (const float*)d_in[1];
  const float* srcmap = (const float*)d_in[2];
  const float* W      = (const float*)d_in[3];
  const float* b      = (const float*)d_in[4];
  const float* Wc     = (const float*)d_in[5];
  const float* bc     = (const float*)d_in[6];

  float* out   = (float*)d_out;
  float* pcopy = out + (size_t)N_ROWS * OUTW;   // second tuple output, flat after first

  pcopy_kernel<<<N_ROWS, 256, 0, stream>>>(hidden, Wc, bc, pcopy);

  dim3 grid(VOCAB / BN, N_ROWS / BM);           // 250 x 16
  gemm_logits_kernel<<<grid, 256, 0, stream>>>(hidden, W, out);

  softmax_scale_kernel<<<N_ROWS, 256, 0, stream>>>(b, pcopy, out);

  copyprob_kernel<<<N_ROWS, 128, 0, stream>>>(attn, srcmap, pcopy, out);
}
